// GraphAttentionalLayer_23055384445457
// MI455X (gfx1250) — matrix operand order repaired and checked by reading, not compiled, not run on hardware
//
#include <hip/hip_runtime.h>

#define NN      6144
#define FIN     256
#define HEADS   4
#define DH      64
#define MASKV   -9.0e15f

typedef __attribute__((ext_vector_type(2)))  float  v2f;
typedef __attribute__((ext_vector_type(8)))  float  v8f;
typedef __attribute__((ext_vector_type(16))) __bf16 v16bf;

__device__ __forceinline__ unsigned short f2bf_raw(float f) {
  unsigned u = __builtin_bit_cast(unsigned, f);
  u += 0x7FFFu + ((u >> 16) & 1u);          // round-to-nearest-even
  return (unsigned short)(u >> 16);
}

__device__ __forceinline__ float lane_bcast(float v, int srcLane) {
  return __builtin_bit_cast(float,
      __builtin_amdgcn_ds_bpermute(srcLane << 2, __builtin_bit_cast(int, v)));
}

// ---------------------------------------------------------------------------
// Kernel 1: ht = h @ W  (fp32 WMMA 16x16x4, keeps score path in full fp32)
// Writes ht_f32 [NN][256] and htT_bf16 [256][NN] (= [head][d][node], V^T).
// ---------------------------------------------------------------------------
__global__ __launch_bounds__(256) void gat_proj(
    const float* __restrict__ h, const float* __restrict__ W,
    float* __restrict__ htf, unsigned short* __restrict__ htT)
{
  const int lane = threadIdx.x & 31;
  const int wid  = (blockIdx.x * (blockDim.x >> 5)) + (threadIdx.x >> 5); // 0..6143
  const int nt = wid & 15, mt = wid >> 4;
  const int m0 = mt * 16, n0 = nt * 16;
  const int h2 = lane >> 4;
  const int nn = lane & 15;

  const float* arow = h + (long)(m0 + nn) * FIN;
  v8f c = {};
  #pragma unroll 8
  for (int k = 0; k < FIN; k += 4) {
    const float2 av = *(const float2*)(arow + k + 2 * h2);
    v2f a; a.x = av.x; a.y = av.y;
    v2f b;
    b.x = W[(long)(k + 2 * h2)     * FIN + n0 + nn];
    b.y = W[(long)(k + 2 * h2 + 1) * FIN + n0 + nn];
    c = __builtin_amdgcn_wmma_f32_16x16x4_f32(false, a, false, b, (short)0, c,
                                              false, false);
  }
  #pragma unroll
  for (int r = 0; r < 8; ++r) {
    const int row = m0 + r + 8 * h2;
    const int col = n0 + nn;
    htf[(long)row * FIN + col] = c[r];
    htT[(long)col * NN + row]  = f2bf_raw(c[r]);
  }
}

// ---------------------------------------------------------------------------
// Kernel 2: src[h][n] = ht[n, h*64..] . a[h,:64], tgt likewise with a[h,64:]
// ---------------------------------------------------------------------------
__global__ __launch_bounds__(256) void gat_scores(
    const float* __restrict__ htf, const float* __restrict__ a,
    float* __restrict__ src, float* __restrict__ tgt)
{
  const int idx = blockIdx.x * blockDim.x + threadIdx.x;
  if (idx >= HEADS * NN) return;
  const int hd = idx / NN;
  const int n  = idx - hd * NN;
  const float* row = htf + (long)n * FIN + hd * DH;
  const float* ah  = a + hd * 2 * DH;
  float s = 0.f, t = 0.f;
  #pragma unroll 8
  for (int d = 0; d < DH; ++d) {
    const float v = row[d];
    s += v * ah[d];
    t += v * ah[DH + d];
  }
  src[hd * NN + n] = s;
  tgt[hd * NN + n] = t;
}

// ---------------------------------------------------------------------------
// Kernel 3 helpers
// ---------------------------------------------------------------------------
// Score one 32-wide sub-chunk: fills ev[16] (A-fragment element order) and
// returns this lane's local max. A 16x32 bf16 layout: lane half h2 owns
// K = 16*h2+{0..15} of row (lane&15).
__device__ __forceinline__ float score_chunk(
    const float* __restrict__ tgtH, const int* __restrict__ adj, long adjRow,
    int j0, int kb, float src_m, float* ev)
{
  alignas(16) float tv[16];
  alignas(16) int   av[16];
  ((float4*)tv)[0] = *(const float4*)(tgtH + j0 + kb);
  ((float4*)tv)[1] = *(const float4*)(tgtH + j0 + kb + 4);
  ((float4*)tv)[2] = *(const float4*)(tgtH + j0 + kb + 8);
  ((float4*)tv)[3] = *(const float4*)(tgtH + j0 + kb + 12);
  ((int4*)av)[0] = *(const int4*)(adj + adjRow + j0 + kb);
  ((int4*)av)[1] = *(const int4*)(adj + adjRow + j0 + kb + 4);
  ((int4*)av)[2] = *(const int4*)(adj + adjRow + j0 + kb + 8);
  ((int4*)av)[3] = *(const int4*)(adj + adjRow + j0 + kb + 12);

  float lmax = -__builtin_inff();
  #pragma unroll
  for (int i = 0; i < 16; ++i) {
    float e = src_m + tv[i];
    e = (e >= 0.f) ? e : 0.01f * e;          // leaky relu
    e = (av[i] > 0) ? e : MASKV;             // adjacency mask
    ev[i] = e;
    lmax = fmaxf(lmax, e);
  }
  return lmax;
}

// ---------------------------------------------------------------------------
// Kernel 3: fused masked-softmax + (attn @ V), flash style.
// One wave owns (head, 16-row tile). 64 columns per online-softmax update
// (two WMMA K-chunks); accumulator rescale is skipped via a wave-uniform
// branch when the running max is unchanged (steady state for dense adj).
// ---------------------------------------------------------------------------
__global__ __launch_bounds__(256) void gat_attn(
    const unsigned short* __restrict__ htT_u,  // [256][NN] bf16 (V^T)
    const float* __restrict__ src,             // [H][NN]
    const float* __restrict__ tgt,             // [H][NN]
    const int*  __restrict__ adj,              // [NN][NN]
    float* __restrict__ out)                   // [NN][256]
{
  const __bf16* htT = (const __bf16*)htT_u;
  const int lane = threadIdx.x & 31;
  const int wid  = (blockIdx.x * (blockDim.x >> 5)) + (threadIdx.x >> 5); // 0..1535
  const int head = wid & 3;
  const int i0   = (wid >> 2) * 16;
  const int h2   = lane >> 4;
  const int nn   = lane & 15;
  const int kb   = h2 * 16;                    // A-side K offset for this half

  const float src_m  = src[head * NN + i0 + nn];
  const float* tgtH  = tgt + head * NN;
  const long  adjRow = (long)(i0 + nn) * NN;

  v8f c0 = {}, c1 = {}, c2 = {}, c3 = {};
  float run_m = -__builtin_inff();
  float run_s = 0.f;

  for (int j0 = 0; j0 < NN; j0 += 64) {
    __builtin_prefetch(adj + adjRow + j0 + 128, 0, 0);   // adj chunk two ahead

    // ---- scores for both 32-wide sub-chunks ----
    float ev0[16], ev1[16];
    float lmax =        score_chunk(tgtH, adj, adjRow, j0,      kb, src_m, ev0);
    lmax = fmaxf(lmax,  score_chunk(tgtH, adj, adjRow, j0 + 32, kb, src_m, ev1));
    lmax = fmaxf(lmax, lane_bcast(lmax, lane ^ 16));     // full row chunk-max

    // ---- online max update (wave-uniform skip when max unchanged) ----
    if (__any(lmax > run_m)) {
      const float new_m = fmaxf(run_m, lmax);
      const float alpha = __expf(run_m - new_m);         // 0 on first chunk
      run_m = new_m;
      run_s *= alpha;
      #pragma unroll
      for (int r = 0; r < 8; ++r) {
        const float ar = lane_bcast(alpha, r + 8 * h2);
        c0[r] *= ar; c1[r] *= ar; c2[r] *= ar; c3[r] *= ar;
      }
    }

    // ---- P = exp(e - m), packed to bf16 A-fragments ----
    v16bf aF0, aF1;
    float ps = 0.f;
    #pragma unroll
    for (int i = 0; i < 16; ++i) {
      const float p = __expf(ev0[i] - run_m);
      ps += p;
      aF0[i] = __builtin_bit_cast(__bf16, f2bf_raw(p));
    }
    #pragma unroll
    for (int i = 0; i < 16; ++i) {
      const float p = __expf(ev1[i] - run_m);
      ps += p;
      aF1[i] = __builtin_bit_cast(__bf16, f2bf_raw(p));
    }
    ps += lane_bcast(ps, lane ^ 16);
    run_s += ps;

    // ---- C += P x V : B 32x16 bf16 = 16 contiguous bf16 per lane ----
    {
      const long base = (long)(head * DH) * NN + j0 + h2 * 16;
      const v16bf b0 = *(const v16bf*)(htT + base + (long)(nn +  0) * NN);
      const v16bf b1 = *(const v16bf*)(htT + base + (long)(nn + 16) * NN);
      const v16bf b2 = *(const v16bf*)(htT + base + (long)(nn + 32) * NN);
      const v16bf b3 = *(const v16bf*)(htT + base + (long)(nn + 48) * NN);
      c0 = __builtin_amdgcn_wmma_f32_16x16x32_bf16(false, aF0, false, b0, (short)0, c0, false, false);
      c1 = __builtin_amdgcn_wmma_f32_16x16x32_bf16(false, aF0, false, b1, (short)0, c1, false, false);
      c2 = __builtin_amdgcn_wmma_f32_16x16x32_bf16(false, aF0, false, b2, (short)0, c2, false, false);
      c3 = __builtin_amdgcn_wmma_f32_16x16x32_bf16(false, aF0, false, b3, (short)0, c3, false, false);
    }
    {
      const long base = (long)(head * DH) * NN + j0 + 32 + h2 * 16;
      const v16bf b0 = *(const v16bf*)(htT + base + (long)(nn +  0) * NN);
      const v16bf b1 = *(const v16bf*)(htT + base + (long)(nn + 16) * NN);
      const v16bf b2 = *(const v16bf*)(htT + base + (long)(nn + 32) * NN);
      const v16bf b3 = *(const v16bf*)(htT + base + (long)(nn + 48) * NN);
      c0 = __builtin_amdgcn_wmma_f32_16x16x32_bf16(false, aF1, false, b0, (short)0, c0, false, false);
      c1 = __builtin_amdgcn_wmma_f32_16x16x32_bf16(false, aF1, false, b1, (short)0, c1, false, false);
      c2 = __builtin_amdgcn_wmma_f32_16x16x32_bf16(false, aF1, false, b2, (short)0, c2, false, false);
      c3 = __builtin_amdgcn_wmma_f32_16x16x32_bf16(false, aF1, false, b3, (short)0, c3, false, false);
    }
  }

  #pragma unroll
  for (int r = 0; r < 8; ++r) {
    const int   rr   = r + 8 * h2;
    const float sInv = 1.f / lane_bcast(run_s, rr);
    float* o = out + (long)(i0 + rr) * FIN + head * DH;
    o[nn +  0] = c0[r] * sInv;
    o[nn + 16] = c1[r] * sInv;
    o[nn + 32] = c2[r] * sInv;
    o[nn + 48] = c3[r] * sInv;
  }
}

// ---------------------------------------------------------------------------
extern "C" void kernel_launch(void* const* d_in, const int* in_sizes, int n_in,
                              void* d_out, int out_size, void* d_ws, size_t ws_size,
                              hipStream_t stream) {
  const float* h   = (const float*)d_in[0];
  const int*   adj = (const int*)d_in[1];
  const float* W   = (const float*)d_in[2];
  const float* a   = (const float*)d_in[3];
  float* out = (float*)d_out;

  char* ws = (char*)d_ws;
  float*          htf = (float*)ws;                                     // 6144*256 f32
  unsigned short* htT = (unsigned short*)(ws + (size_t)NN * FIN * 4);   // 256*6144 bf16
  float*          src = (float*)(ws + (size_t)NN * FIN * 4 + (size_t)NN * FIN * 2);
  float*          tgt = src + HEADS * NN;                               // 2 * 4*6144 f32

  gat_proj<<<768, 256, 0, stream>>>(h, W, htf, htT);
  gat_scores<<<(HEADS * NN + 255) / 256, 256, 0, stream>>>(htf, a, src, tgt);
  gat_attn<<<192, 256, 0, stream>>>(htT, src, tgt, adj, out);
}
